// Transformer_Blocks_45389214384227
// MI455X (gfx1250) — matrix operand order repaired and checked by reading, not compiled, not run on hardware
//
#include <hip/hip_runtime.h>
#include <cmath>

// ---------------- problem constants ----------------
constexpr int cB  = 32;
constexpr int cT  = 512;
constexpr int cD  = 512;
constexpr int cHD = 64;
constexpr int cHN = 16;
constexpr int cO  = 512;
constexpr int cBT = cB * cT;          // 16384
constexpr int cCC = cHN * cHD;        // 1024 (concat width)

typedef _Float16 v16h __attribute__((ext_vector_type(16)));
typedef _Float16 v8h  __attribute__((ext_vector_type(8)));
typedef float    v8f  __attribute__((ext_vector_type(8)));

// ---------------- async global->LDS copy (gfx1250 GLOBAL_LOAD_ASYNC_TO_LDS_B128) ----------------
// The clang builtin __builtin_amdgcn_global_load_lds is not declared on this toolchain
// (verified via __has_builtin in round 2), so issue the CDNA5 instruction directly.
// LDS byte address = low 32 bits of the flat pointer (ISA 10.2: LDS aperture keeps
// the offset in addr[31:0]).
#ifndef USE_ASYNC_ASM
#define USE_ASYNC_ASM 1
#endif

__device__ __forceinline__ void cp16(const _Float16* g, _Float16* s) {
#if USE_ASYNC_ASM
  unsigned lds = (unsigned)(uintptr_t)s;
  unsigned long long ga = (unsigned long long)(uintptr_t)g;
  asm volatile("global_load_async_to_lds_b128 %0, %1, off"
               :: "v"(lds), "v"(ga) : "memory");
#else
  *(v8h*)s = *(const v8h*)g;   // fallback: load + ds_store
#endif
}
__device__ __forceinline__ void async_fence() {
#if USE_ASYNC_ASM
  asm volatile("s_wait_asynccnt 0" ::: "memory");
#endif
}

// ---------------- fragment helpers (layouts per CDNA5 ISA) ----------------
__device__ __forceinline__ v16h cat8(v8h a, v8h b) {
  return __builtin_shufflevector(a, b, 0,1,2,3,4,5,6,7,8,9,10,11,12,13,14,15);
}
// B fragment: 16 contiguous f16 (K = base .. base+15 for this half-wave)
__device__ __forceinline__ v16h load16h(const _Float16* p) {
  v8h a = *(const v8h*)p;
  v8h b = *(const v8h*)(p + 8);
  return cat8(a, b);
}
// A fragment: elements 0..15 = K 2*kbase..2*kbase+15 (contiguous, matches B)
__device__ __forceinline__ v16h load_afrag(const _Float16* rowp, int kbase) {
  v8h a = *(const v8h*)(rowp + 2 * kbase);
  v8h b = *(const v8h*)(rowp + 2 * kbase + 8);
  return cat8(a, b);
}
__device__ __forceinline__ v8f wmma32(v16h a, v16h b, v8f c) {
  return __builtin_amdgcn_wmma_f32_16x16x32_f16(false, a, false, b, (short)0, c, false, false);
}

// ---------------- elementwise f32 -> f16 ----------------
__global__ void k_cvt(const float* __restrict__ s, _Float16* __restrict__ d, long n) {
  long i = (long)blockIdx.x * blockDim.x + threadIdx.x;
  if (i < n) d[i] = (_Float16)s[i];
}

// transpose+convert: src f32 [K][N] row-major -> dst f16 [N][K]; batch via blockIdx.y
__global__ void k_transpose_cvt(const float* __restrict__ src, _Float16* __restrict__ dst,
                                int K, int N) {
  long z = blockIdx.y;
  src += z * (long)K * N;
  dst += z * (long)K * N;
  long idx = (long)blockIdx.x * blockDim.x + threadIdx.x;
  if (idx < (long)K * N) {
    int k = (int)(idx / N), n = (int)(idx % N);
    dst[(long)n * K + k] = (_Float16)src[idx];
  }
}

// ---------------- generic WMMA GEMM, LDS double-buffered, async staged ----------------
// C[z] = act(A[z] (f16,[M,K]) @ B[z] (f16 pre-transposed [N,K]) + bias[z])
// block = 128 threads (4 waves); block tile 128(M) x 64(N); wave tile 32 x 64; BK=32.
// act: 0 none, 1 relu, 2 tanh.  transC: store C[n*ldc+m] (f16).  outF16: f16 vs f32 output.
__global__ __launch_bounds__(128) void k_gemm(
    const _Float16* __restrict__ A, const _Float16* __restrict__ BT,
    const float* __restrict__ bias, void* __restrict__ Cout,
    int M, int N, int K, int lda, int ldc,
    long aStride, int aDiv, long bStride, int bMod, int biasStride,
    long cStride, int act, int transC, int outF16)
{
  __shared__ _Float16 sA[2][128][40];  // [buf][row m][K 0..31], stride 40 keeps 16B align
  __shared__ _Float16 sB[2][64][40];   // [buf][row n][K 0..31]

  int z = blockIdx.z;
  const _Float16* Az    = A  + (long)(z / aDiv) * aStride + (long)blockIdx.y * 128 * lda;
  const _Float16* Bz    = BT + (long)(z % bMod) * bStride + (long)blockIdx.x * 64 * K;
  const float*    biasz = bias + (long)(z % bMod) * biasStride;

  int tid  = threadIdx.x;
  int wave = tid >> 5;
  int lane = tid & 31;
  int lh = lane >> 4, lm = lane & 15;
  int kbase = lh * 8;
  int n0 = blockIdx.x * 64;

  // staging: A 128x32 (8KB) + B 64x32 (4KB) per buffer = 768 x 16B chunks / 128 threads
  int srow = tid >> 2;                 // 0..31
  int sseg = (tid & 3) * 8;            // halves

  auto stage = [&](int k0, int buf) {
    const _Float16* ga = Az + (long)srow * lda + k0 + sseg;
#pragma unroll
    for (int i = 0; i < 4; i++)
      cp16(ga + (long)i * 32 * lda, &sA[buf][srow + i * 32][sseg]);
    const _Float16* gb = Bz + (long)srow * K + k0 + sseg;
    cp16(gb,                &sB[buf][srow][sseg]);
    cp16(gb + 32 * (long)K, &sB[buf][srow + 32][sseg]);
  };

  v8f acc0[4] = {}, acc1[4] = {};
  int nsteps = K >> 5;

  stage(0, 0);
  for (int s = 0; s < nsteps; ++s) {
    int cur = s & 1;
    async_fence();          // own async copies into `cur` complete
    __syncthreads();        // everyone's copies complete; prev compute done -> cur^1 reusable
    if (s + 1 < nsteps) stage((s + 1) << 5, cur ^ 1);

    v16h a0 = load_afrag(&sA[cur][wave * 32 + lm][0],      kbase);
    v16h a1 = load_afrag(&sA[cur][wave * 32 + 16 + lm][0], kbase);
    v16h bfr[4];
#pragma unroll
    for (int t = 0; t < 4; t++) bfr[t] = load16h(&sB[cur][t * 16 + lm][lh * 16]);
#pragma unroll
    for (int t = 0; t < 4; t++) {
      acc0[t] = wmma32(a0, bfr[t], acc0[t]);
      acc1[t] = wmma32(a1, bfr[t], acc1[t]);
    }
  }

  long cOff = (long)z * cStride;
  int mw = blockIdx.y * 128 + wave * 32;

  auto epil = [&](v8f* acc, int m0) {
#pragma unroll
    for (int t = 0; t < 4; t++) {
      int n = n0 + t * 16 + lm;
      float bv = biasz[n];
      float vals[8];
#pragma unroll
      for (int r = 0; r < 8; r++) {
        float x = acc[t][r] + bv;
        if (act == 1)      x = fmaxf(x, 0.0f);
        else if (act == 2) x = tanhf(x);
        vals[r] = x;
      }
      if (transC) {
        _Float16* dst = (_Float16*)Cout + cOff + (long)n * ldc + m0 + lh * 8;
        v8h h;
#pragma unroll
        for (int r = 0; r < 8; r++) h[r] = (_Float16)vals[r];
        *(v8h*)dst = h;                       // rows contiguous in transposed layout
      } else if (outF16) {
        _Float16* dst = (_Float16*)Cout + cOff;
#pragma unroll
        for (int r = 0; r < 8; r++)
          dst[(long)(m0 + lh * 8 + r) * ldc + n] = (_Float16)vals[r];
      } else {
        float* dst = (float*)Cout + cOff;
#pragma unroll
        for (int r = 0; r < 8; r++)
          dst[(long)(m0 + lh * 8 + r) * ldc + n] = vals[r];
      }
    }
  };
  epil(acc0, mw);
  epil(acc1, mw + 16);
}

// ---------------- flash attention (software-pipelined) ----------------
// q,k: f16 [B*HN][T][HD]; vT: f16 [B*HN][HD][T]; out: concat f16 [B][T][HN*HD]
// block = 256 (8 waves); wave handles one 16-row t-tile; grid = (T/128, B*HN)
__global__ __launch_bounds__(256) void k_attn(
    const _Float16* __restrict__ q, const _Float16* __restrict__ k,
    const _Float16* __restrict__ vT, _Float16* __restrict__ outc)
{
  __shared__ _Float16 pbuf[8][16 * 40];   // per-wave P tile [16 t][32 s], row stride 40

  int wave = threadIdx.x >> 5;
  int lane = threadIdx.x & 31;
  int lh = lane >> 4, lm = lane & 15;
  int kbase = lh * 8;
  int z  = blockIdx.y;                         // b*HN + h
  int t0 = (blockIdx.x * 8 + wave) * 16;

  const _Float16* qz = q  + (long)z * cT * cHD;
  const _Float16* kz = k  + (long)z * cT * cHD;
  const _Float16* vz = vT + (long)z * cHD * cT;

  const _Float16* qrow = qz + (long)(t0 + lm) * cHD;
  v16h aq0 = load_afrag(qrow,      kbase);     // f = 0..31
  v16h aq1 = load_afrag(qrow + 32, kbase);     // f = 32..63

  v8f acc[4] = {};
  float mrun[8], lrun[8];
#pragma unroll
  for (int r = 0; r < 8; r++) { mrun[r] = -INFINITY; lrun[r] = 0.0f; }

  _Float16* pb = &pbuf[wave][0];
  int tmax = t0 + 15;

  // preload K-fragments of the first chunk
  v16h kf0a, kf0b, kf1a, kf1b;
  {
    const _Float16* kr0 = kz + (long)lm * cHD;
    const _Float16* kr1 = kz + (long)(16 + lm) * cHD;
    kf0a = load16h(kr0 + lh * 16); kf0b = load16h(kr0 + 32 + lh * 16);
    kf1a = load16h(kr1 + lh * 16); kf1b = load16h(kr1 + 32 + lh * 16);
  }

  for (int s0 = 0; s0 <= tmax; s0 += 32) {     // causal skip of fully-masked chunks
    // ---- S = (q @ k^T) * 1/sqrt(HD), two 16x16 tiles ----
    v8f st[2];
    { v8f sc = {}; sc = wmma32(aq0, kf0a, sc); sc = wmma32(aq1, kf0b, sc); st[0] = sc * 0.125f; }
    { v8f sc = {}; sc = wmma32(aq0, kf1a, sc); sc = wmma32(aq1, kf1b, sc); st[1] = sc * 0.125f; }

    // ---- prefetch next chunk's K-fragments (long latency window) ----
    int sn = s0 + 32;
    if (sn <= tmax) {
      const _Float16* kr0 = kz + (long)(sn + lm) * cHD;
      const _Float16* kr1 = kz + (long)(sn + 16 + lm) * cHD;
      kf0a = load16h(kr0 + lh * 16); kf0b = load16h(kr0 + 32 + lh * 16);
      kf1a = load16h(kr1 + lh * 16); kf1b = load16h(kr1 + 32 + lh * 16);
    }
    // ---- prefetch V-fragments for the current chunk ----
    v16h bb[4];
#pragma unroll
    for (int c = 0; c < 4; c++)
      bb[c] = load16h(vz + (long)(c * 16 + lm) * cT + s0 + lh * 16);

    // ---- causal mask + online softmax update (fp32) ----
    float pv[2][8], mnew[8], alpha[8];
#pragma unroll
    for (int r = 0; r < 8; r++) {
      int trow = t0 + lh * 8 + r;
#pragma unroll
      for (int j = 0; j < 2; j++) {
        int scol = s0 + j * 16 + lm;
        float x = st[j][r];
        if (scol > trow) x = -1e9f;            // matches reference masking
        pv[j][r] = x;
      }
      float mx = fmaxf(pv[0][r], pv[1][r]);
#pragma unroll
      for (int dlt = 1; dlt < 16; dlt <<= 1) mx = fmaxf(mx, __shfl_xor(mx, dlt, 32));
      mnew[r]  = fmaxf(mrun[r], mx);
      alpha[r] = __expf(mrun[r] - mnew[r]);
      float p0 = __expf(pv[0][r] - mnew[r]);
      float p1 = __expf(pv[1][r] - mnew[r]);
      pv[0][r] = p0; pv[1][r] = p1;
      float rs = p0 + p1;
#pragma unroll
      for (int dlt = 1; dlt < 16; dlt <<= 1) rs += __shfl_xor(rs, dlt, 32);
      lrun[r] = lrun[r] * alpha[r] + rs;
      mrun[r] = mnew[r];
    }
#pragma unroll
    for (int c = 0; c < 4; c++)
#pragma unroll
      for (int r = 0; r < 8; r++) acc[c][r] *= alpha[r];

    // ---- re-swizzle P (C-layout -> A-layout) through per-wave LDS ----
#pragma unroll
    for (int r = 0; r < 8; r++) {
      int row = lh * 8 + r;
      pb[row * 40 + lm]      = (_Float16)pv[0][r];
      pb[row * 40 + 16 + lm] = (_Float16)pv[1][r];
    }
    __builtin_amdgcn_wave_barrier();
    asm volatile("s_wait_dscnt 0" ::: "memory");   // DS ops are in-order per wave
    v16h ap = load_afrag(pb + lm * 40, kbase);     // A: 16(t) x 32(s)
    __builtin_amdgcn_wave_barrier();

    // ---- O += P @ V with prefetched V fragments ----
#pragma unroll
    for (int c = 0; c < 4; c++) acc[c] = wmma32(ap, bb[c], acc[c]);
  }

  // ---- epilogue: normalize and write concat (head-major columns) ----
  int b = z >> 4, h = z & 15;
#pragma unroll
  for (int c = 0; c < 4; c++)
#pragma unroll
    for (int r = 0; r < 8; r++) {
      int trow = t0 + lh * 8 + r;
      float o = acc[c][r] / lrun[r];
      outc[((long)(b * cT + trow)) * cCC + h * cHD + c * 16 + lm] = (_Float16)o;
    }
}

// ---------------- fused add + layernorm ----------------
// y = LN(x + res)*w + b ; optionally also emit f16 copy. One wave per 512-wide row.
__global__ __launch_bounds__(256) void k_add_ln(
    const float* __restrict__ x, const float* __restrict__ res,
    const float* __restrict__ w, const float* __restrict__ b,
    float* __restrict__ y, _Float16* __restrict__ yh)
{
  int wave = threadIdx.x >> 5, lane = threadIdx.x & 31;
  long row = (long)blockIdx.x * 8 + wave;
  const float* xr = x  + row * cO;
  const float* rr = res + row * cO;
  float v[16];
  float s = 0.0f;
#pragma unroll
  for (int i = 0; i < 16; i++) { int c = lane + i * 32; v[i] = xr[c] + rr[c]; s += v[i]; }
#pragma unroll
  for (int d = 1; d < 32; d <<= 1) s += __shfl_xor(s, d, 32);
  float mu = s * (1.0f / cO);
  float s2 = 0.0f;
#pragma unroll
  for (int i = 0; i < 16; i++) { float dd = v[i] - mu; s2 += dd * dd; }
#pragma unroll
  for (int d = 1; d < 32; d <<= 1) s2 += __shfl_xor(s2, d, 32);
  float inv = rsqrtf(s2 * (1.0f / cO) + 1e-5f);
#pragma unroll
  for (int i = 0; i < 16; i++) {
    int c = lane + i * 32;
    float o = (v[i] - mu) * inv * w[c] + b[c];
    y[row * cO + c] = o;
    if (yh) yh[row * cO + c] = (_Float16)o;
  }
}

// ---------------- host orchestration ----------------
extern "C" void kernel_launch(void* const* d_in, const int* in_sizes, int n_in,
                              void* d_out, int out_size, void* d_ws, size_t ws_size,
                              hipStream_t stream) {
  const float* inp  = (const float*)d_in[0];
  const float* Wq   = (const float*)d_in[1];
  const float* bq   = (const float*)d_in[2];
  const float* Wk   = (const float*)d_in[3];
  const float* bk   = (const float*)d_in[4];
  const float* Wv   = (const float*)d_in[5];
  const float* bv   = (const float*)d_in[6];
  const float* Wr   = (const float*)d_in[7];
  const float* br   = (const float*)d_in[8];
  const float* Wo   = (const float*)d_in[9];
  const float* bo   = (const float*)d_in[10];
  const float* ln1w = (const float*)d_in[11];
  const float* ln1b = (const float*)d_in[12];
  const float* Wfc  = (const float*)d_in[13];
  const float* bfc  = (const float*)d_in[14];
  const float* ln2w = (const float*)d_in[15];
  const float* ln2b = (const float*)d_in[16];

  char* ws = (char*)d_ws;
  size_t cur = 0;
  auto alloc = [&](size_t bytes) { size_t r = cur; cur += (bytes + 255) & ~(size_t)255; return r; };

  _Float16* X16   = (_Float16*)(ws + alloc((size_t)cBT * cD * 2));        // 16 MB
  _Float16* WrT   = (_Float16*)(ws + alloc((size_t)cD * cO * 2));
  _Float16* WqT   = (_Float16*)(ws + alloc((size_t)cHN * cD * cHD * 2));
  _Float16* WkT   = (_Float16*)(ws + alloc((size_t)cHN * cD * cHD * 2));
  _Float16* WvT   = (_Float16*)(ws + alloc((size_t)cHN * cD * cHD * 2));
  _Float16* WoT   = (_Float16*)(ws + alloc((size_t)cCC * cO * 2));
  _Float16* WfcT  = (_Float16*)(ws + alloc((size_t)cO * cO * 2));
  _Float16* q16   = (_Float16*)(ws + alloc((size_t)cB * cHN * cT * cHD * 2));  // 32 MB
  _Float16* k16   = (_Float16*)(ws + alloc((size_t)cB * cHN * cT * cHD * 2));
  _Float16* vT16  = (_Float16*)(ws + alloc((size_t)cB * cHN * cHD * cT * 2));
  _Float16* cc16  = (_Float16*)(ws + alloc((size_t)cBT * cCC * 2));            // 32 MB
  float*    resv  = (float*)   (ws + alloc((size_t)cBT * cO * 4));             // 32 MB
  // overlays (lifetimes are disjoint):
  float*    fcv = (float*)q16;      // FC output (after attention; q dead)
  float*    x1  = (float*)k16;      // Wo projection output (k dead)
  float*    y1  = (float*)vT16;     // ln1 output f32 (vT dead)
  _Float16* y1h = (_Float16*)cc16;  // ln1 output f16 (concat dead after Wo gemm)

  // 1) activations -> f16
  {
    long n = (long)cBT * cD;
    k_cvt<<<dim3((unsigned)((n + 255) / 256)), dim3(256), 0, stream>>>(inp, X16, n);
  }
  // 2) weights -> transposed f16 [N][K]
  k_transpose_cvt<<<dim3((cD * cO + 255) / 256, 1),    dim3(256), 0, stream>>>(Wr,  WrT,  cD,  cO);
  k_transpose_cvt<<<dim3((cD * cHD + 255) / 256, cHN), dim3(256), 0, stream>>>(Wq,  WqT,  cD,  cHD);
  k_transpose_cvt<<<dim3((cD * cHD + 255) / 256, cHN), dim3(256), 0, stream>>>(Wk,  WkT,  cD,  cHD);
  k_transpose_cvt<<<dim3((cD * cHD + 255) / 256, cHN), dim3(256), 0, stream>>>(Wv,  WvT,  cD,  cHD);
  k_transpose_cvt<<<dim3((cCC * cO + 255) / 256, 1),   dim3(256), 0, stream>>>(Wo,  WoT,  cCC, cO);
  k_transpose_cvt<<<dim3((cO * cO + 255) / 256, 1),    dim3(256), 0, stream>>>(Wfc, WfcT, cO,  cO);

  // 3) residual = X @ Wr + br            (f32 out)
  k_gemm<<<dim3(cO / 64, cT / 128, cB), dim3(128), 0, stream>>>(
      X16, WrT, br, resv, cT, cO, cD, cD, cO,
      (long)cT * cD, 1, 0L, 1, 0, (long)cT * cO, 0, 0, 0);
  // 4) q = relu(X @ Wq + bq)             (f16 out, [z][t][f])
  k_gemm<<<dim3(1, cT / 128, cB * cHN), dim3(128), 0, stream>>>(
      X16, WqT, bq, q16, cT, cHD, cD, cD, cHD,
      (long)cT * cD, cHN, (long)cD * cHD, cHN, cHD, (long)cT * cHD, 1, 0, 1);
  // 5) k = relu(X @ Wk + bk)
  k_gemm<<<dim3(1, cT / 128, cB * cHN), dim3(128), 0, stream>>>(
      X16, WkT, bk, k16, cT, cHD, cD, cD, cHD,
      (long)cT * cD, cHN, (long)cD * cHD, cHN, cHD, (long)cT * cHD, 1, 0, 1);
  // 6) v = tanh(X @ Wv + bv), stored transposed [z][f][t]
  k_gemm<<<dim3(1, cT / 128, cB * cHN), dim3(128), 0, stream>>>(
      X16, WvT, bv, vT16, cT, cHD, cD, cD, cT,
      (long)cT * cD, cHN, (long)cD * cHD, cHN, cHD, (long)cHD * cT, 2, 1, 1);
  // 7) flash attention -> concat f16 [B][T][HN*HD]
  k_attn<<<dim3(cT / 128, cB * cHN), dim3(256), 0, stream>>>(q16, k16, vT16, cc16);
  // 8) x1 = tanh(concat @ Wo + bo)       (f32 out)
  k_gemm<<<dim3(cO / 64, cT / 128, cB), dim3(128), 0, stream>>>(
      cc16, WoT, bo, x1, cT, cO, cCC, cCC, cO,
      (long)cT * cCC, 1, 0L, 1, 0, (long)cT * cO, 2, 0, 0);
  // 9) y1 = LN(x1 + residual)            (f32 + f16)
  k_add_ln<<<dim3(cBT / 8), dim3(256), 0, stream>>>(x1, resv, ln1w, ln1b, y1, y1h);
  // 10) fc = relu(y1 @ Wfc + bfc)        (f32 out)
  k_gemm<<<dim3(cO / 64, cT / 128, cB), dim3(128), 0, stream>>>(
      y1h, WfcT, bfc, fcv, cT, cO, cO, cO, cO,
      (long)cT * cO, 1, 0L, 1, 0, (long)cT * cO, 1, 0, 0);
  // 11) out = LN(fc + y1)                (f32 -> d_out)
  k_add_ln<<<dim3(cBT / 8), dim3(256), 0, stream>>>(fcv, y1, ln2w, ln2b, (float*)d_out, nullptr);
}
